// MambaLM_87359634801439
// MI455X (gfx1250) — matrix operand order repaired and checked by reading, not compiled, not run on hardware
//
#include <hip/hip_runtime.h>
#include <hip/hip_bf16.h>

// Problem constants (match reference)
#define V_ 32000
#define D_ 1024
#define L_ 4
#define T_ 1024
#define B_ 2
#define E_ 2048      // d_inner
#define N_ 16        // d_state
#define R_ 64        // dt_rank
#define K_ 4         // d_conv
#define M_ (B_*T_)   // 2048 rows

typedef __attribute__((ext_vector_type(16))) __bf16 v16bf;
typedef __attribute__((ext_vector_type(8)))  float  v8f;
typedef __attribute__((ext_vector_type(4)))  int    v4i_;

union FragB { uint4 u[2]; v16bf v; };

__device__ inline __bf16 f2bf(float f) {
    unsigned u = __float_as_uint(f);
    unsigned r = (u + 0x7FFFu + ((u >> 16) & 1u)) >> 16;   // round-to-nearest-even
    unsigned short s = (unsigned short)r;
    __bf16 b; __builtin_memcpy(&b, &s, 2); return b;
}

__device__ inline v8f v8f_zero() {
    v8f z;
#pragma unroll
    for (int i = 0; i < 8; ++i) z[i] = 0.f;
    return z;
}

// --- CDNA5 async global->LDS copy (ASYNCcnt path), guarded for portability ---
#if __has_builtin(__builtin_amdgcn_global_load_async_to_lds_b128)
#define ASYNC_LDS 1
typedef __attribute__((address_space(1))) v4i_* gv4p;
typedef __attribute__((address_space(3))) v4i_* lv4p;
__device__ inline void async_cp16(const __bf16* g, __bf16* l) {
    __builtin_amdgcn_global_load_async_to_lds_b128((gv4p)g, (lv4p)l, 0, 0);
}
__device__ inline void async_wait0() {
#if __has_builtin(__builtin_amdgcn_s_wait_asynccnt)
    __builtin_amdgcn_s_wait_asynccnt(0);
#else
    asm volatile("s_wait_asynccnt 0x0" ::: "memory");
#endif
}
#else
#define ASYNC_LDS 0
#endif

// ---------------------------------------------------------------------------
// Tiled bf16 WMMA GEMM:  C[M,N] (f32) = A[M,K] (bf16, row-major) @ B[N,K]^T
// Block: 256 threads (8 waves), tile 128x256, BK=32 (one v_wmma per K step).
// Wave tile 64x64 -> 4x4 f32 16x16 accumulators (1.0 ds_load_b128 per wmma).
// M, K assumed multiples of 128 / 32; N bounds-checked (handles N=96).
// ---------------------------------------------------------------------------
#define BM 128
#define BN 256
#define BK 32
#define LDK 40   // padded LDS pitch in bf16 elements (80 B, 16 B aligned)

__global__ __launch_bounds__(256) void k_gemm_bf16(
    float* __restrict__ C, const __bf16* __restrict__ A, const __bf16* __restrict__ B,
    int Mdim, int Ndim, int Kdim)
{
    __shared__ __align__(16) __bf16 At[BM * LDK];   // 10 KB
    __shared__ __align__(16) __bf16 Bt[BN * LDK];   // 20 KB

    const int tid  = threadIdx.x;
    const int bm   = blockIdx.y * BM;
    const int bn   = blockIdx.x * BN;
    const int wid  = tid >> 5;
    const int lane = tid & 31;
    const int wm   = (wid & 1) * 64;   // wave M offset in tile (2 waves in M)
    const int wn   = (wid >> 1) * 64;  // wave N offset in tile (4 waves in N)
    const int lrow = tid >> 1;         // 0..127 (loader row)
    const int lcol = (tid & 1) * 16;   // 0 or 16 (loader col)
    const int half = lane >> 4;
    const int l16  = lane & 15;

    v8f acc[4][4];
#pragma unroll
    for (int mi = 0; mi < 4; ++mi)
#pragma unroll
        for (int ni = 0; ni < 4; ++ni) acc[mi][ni] = v8f_zero();

    const int nB0 = bn + lrow;          // B tile rows handled by this thread
    const int nB1 = bn + 128 + lrow;

    for (int k0 = 0; k0 < Kdim; k0 += BK) {
#if ASYNC_LDS
        // A tile: rows always in-bounds (M multiple of 128)
        {
            const __bf16* src = A + (size_t)(bm + lrow) * Kdim + k0 + lcol;
            async_cp16(src,     &At[lrow * LDK + lcol]);
            async_cp16(src + 8, &At[lrow * LDK + lcol + 8]);
        }
        // B tile: 256 rows, 2 per thread; ragged N zero-filled via plain LDS stores
        {
            uint4 zz = make_uint4(0u, 0u, 0u, 0u);
            if (nB0 < Ndim) {
                const __bf16* src = B + (size_t)nB0 * Kdim + k0 + lcol;
                async_cp16(src,     &Bt[lrow * LDK + lcol]);
                async_cp16(src + 8, &Bt[lrow * LDK + lcol + 8]);
            } else {
                *(uint4*)(&Bt[lrow * LDK + lcol])     = zz;
                *(uint4*)(&Bt[lrow * LDK + lcol + 8]) = zz;
            }
            if (nB1 < Ndim) {
                const __bf16* src = B + (size_t)nB1 * Kdim + k0 + lcol;
                async_cp16(src,     &Bt[(128 + lrow) * LDK + lcol]);
                async_cp16(src + 8, &Bt[(128 + lrow) * LDK + lcol + 8]);
            } else {
                *(uint4*)(&Bt[(128 + lrow) * LDK + lcol])     = zz;
                *(uint4*)(&Bt[(128 + lrow) * LDK + lcol + 8]) = zz;
            }
        }
        async_wait0();
        __syncthreads();
#else
        // Synchronous staging fallback
        {
            const __bf16* src = A + (size_t)(bm + lrow) * Kdim + k0 + lcol;
            uint4 u0 = *(const uint4*)(src);
            uint4 u1 = *(const uint4*)(src + 8);
            *(uint4*)(&At[lrow * LDK + lcol])     = u0;
            *(uint4*)(&At[lrow * LDK + lcol + 8]) = u1;
        }
        {
            uint4 u0 = make_uint4(0u, 0u, 0u, 0u), u1 = u0;
            if (nB0 < Ndim) {
                const __bf16* src = B + (size_t)nB0 * Kdim + k0 + lcol;
                u0 = *(const uint4*)(src);
                u1 = *(const uint4*)(src + 8);
            }
            *(uint4*)(&Bt[lrow * LDK + lcol])     = u0;
            *(uint4*)(&Bt[lrow * LDK + lcol + 8]) = u1;
            u0 = make_uint4(0u, 0u, 0u, 0u); u1 = u0;
            if (nB1 < Ndim) {
                const __bf16* src = B + (size_t)nB1 * Kdim + k0 + lcol;
                u0 = *(const uint4*)(src);
                u1 = *(const uint4*)(src + 8);
            }
            *(uint4*)(&Bt[(128 + lrow) * LDK + lcol])     = u0;
            *(uint4*)(&Bt[(128 + lrow) * LDK + lcol + 8]) = u1;
        }
        __syncthreads();
#endif

        // Prefetch next K tile into cache while we compute (global_prefetch_b8)
        if (k0 + BK < Kdim) {
            __builtin_prefetch(A + (size_t)(bm + lrow) * Kdim + k0 + BK + lcol, 0, 3);
            if (nB0 < Ndim)
                __builtin_prefetch(B + (size_t)nB0 * Kdim + k0 + BK + lcol, 0, 3);
        }

        // A fragments: lane m=l16; k runs {half*8..+7} and {16+half*8..+7}
        FragB afr[4];
#pragma unroll
        for (int mi = 0; mi < 4; ++mi) {
            const __bf16* ar = &At[(wm + mi * 16 + l16) * LDK];
            afr[mi].u[0] = *(const uint4*)(ar + half * 8);
            afr[mi].u[1] = *(const uint4*)(ar + 16 + half * 8);
        }
#pragma unroll
        for (int ni = 0; ni < 4; ++ni) {
            // B fragment: lane n=l16; k runs {half*8..+7} and {16+half*8..+7}
            FragB bfr;
            const __bf16* br = &Bt[(wn + ni * 16 + l16) * LDK];
            bfr.u[0] = *(const uint4*)(br + half * 8);
            bfr.u[1] = *(const uint4*)(br + 16 + half * 8);
#pragma unroll
            for (int mi = 0; mi < 4; ++mi) {
                acc[mi][ni] = __builtin_amdgcn_wmma_f32_16x16x32_bf16(
                    false, afr[mi].v, false, bfr.v, (short)0, acc[mi][ni], false, false);
            }
        }
        __syncthreads();
    }

    // Store: VGPR r holds (M = half*8 + r, N = l16)
#pragma unroll
    for (int mi = 0; mi < 4; ++mi) {
#pragma unroll
        for (int ni = 0; ni < 4; ++ni) {
            int m0 = bm + wm + mi * 16 + half * 8;
            int n  = bn + wn + ni * 16 + l16;
            if (n < Ndim) {
#pragma unroll
                for (int r = 0; r < 8; ++r)
                    C[(size_t)(m0 + r) * Ndim + n] = acc[mi][ni][r];
            }
        }
    }
}

// ---------------------------------------------------------------------------
// Elementwise / scan / reduction kernels
// ---------------------------------------------------------------------------
__global__ void k_cvt_bf16(__bf16* __restrict__ dst, const float* __restrict__ src, int n) {
    int i = blockIdx.x * blockDim.x + threadIdx.x;
    if (i < n) dst[i] = f2bf(src[i]);
}

__global__ void k_embed(float* __restrict__ x, __bf16* __restrict__ xb,
                        const int* __restrict__ ids,
                        const float* __restrict__ tok, const float* __restrict__ pos) {
    int i = blockIdx.x * blockDim.x + threadIdx.x;
    if (i >= B_ * T_ * D_) return;
    int d = i % D_;
    int r = i / D_;          // b*T + t
    int t = r % T_;
    int id = ids[r];
    float v = tok[(size_t)id * D_ + d] + pos[(size_t)t * D_ + d];
    x[i]  = v;
    xb[i] = f2bf(v);
}

// causal depthwise conv (K=4) + bias + SiLU on xz[:, :, 0:E]
__global__ void k_conv_silu(float* __restrict__ xc, __bf16* __restrict__ xcb,
                            const float* __restrict__ xz,
                            const float* __restrict__ cw, const float* __restrict__ cb) {
    int i = blockIdx.x * blockDim.x + threadIdx.x;
    if (i >= M_ * E_) return;
    int e = i % E_;
    int row = i / E_;
    int b = row / T_, t = row % T_;
    float s = cb[e];
#pragma unroll
    for (int k = 0; k < K_; ++k) {
        int tt = t - (K_ - 1) + k;
        if (tt >= 0)
            s += cw[e * K_ + k] * xz[(size_t)(b * T_ + tt) * (2 * E_) + e];
    }
    float sil = s / (1.f + __expf(-s));
    xc[i]  = sil;
    xcb[i] = f2bf(sil);
}

__global__ void k_extract_dt(__bf16* __restrict__ dtb16, const float* __restrict__ xdbl) {
    int i = blockIdx.x * blockDim.x + threadIdx.x;
    if (i >= M_ * R_) return;
    int row = i / R_, r = i % R_;
    dtb16[i] = f2bf(xdbl[(size_t)row * (R_ + 2 * N_) + r]);
}

// selective scan: one thread per (b,e); 16-state recurrence in registers
__global__ void k_scan(float* __restrict__ y, const float* __restrict__ dpre,
                       const float* __restrict__ xdbl, const float* __restrict__ xc,
                       const float* __restrict__ Alog, const float* __restrict__ dtb) {
    int idx = blockIdx.x * blockDim.x + threadIdx.x;
    if (idx >= B_ * E_) return;
    int b = idx / E_, e = idx % E_;
    float A[N_], h[N_];
#pragma unroll
    for (int n = 0; n < N_; ++n) { A[n] = -__expf(Alog[e * N_ + n]); h[n] = 0.f; }
    float bias = dtb[e];
    for (int t = 0; t < T_; ++t) {
        int row = b * T_ + t;
        float dp = dpre[(size_t)row * E_ + e] + bias;
        float delta = (dp > 20.f) ? dp : log1pf(__expf(dp));
        float xv = xc[(size_t)row * E_ + e];
        const float* Bm = xdbl + (size_t)row * (R_ + 2 * N_) + R_;
        const float* Cm = Bm + N_;
        float dx = delta * xv;
        float acc = 0.f;
#pragma unroll
        for (int n = 0; n < N_; ++n) {
            float dA = __expf(delta * A[n]);
            h[n] = dA * h[n] + dx * Bm[n];
            acc += h[n] * Cm[n];
        }
        y[(size_t)row * E_ + e] = acc;
    }
}

// y = (ycore + D*xc) * silu(z); emit bf16 for out_proj GEMM
__global__ void k_gate(__bf16* __restrict__ yb, const float* __restrict__ ycore,
                       const float* __restrict__ xc, const float* __restrict__ xz,
                       const float* __restrict__ Dp) {
    int i = blockIdx.x * blockDim.x + threadIdx.x;
    if (i >= M_ * E_) return;
    int e = i % E_;
    int row = i / E_;
    float z = xz[(size_t)row * (2 * E_) + E_ + e];
    float g = z / (1.f + __expf(-z));
    yb[i] = f2bf((ycore[i] + Dp[e] * xc[i]) * g);
}

__global__ void k_residual(float* __restrict__ x, __bf16* __restrict__ xb,
                           const float* __restrict__ o) {
    int i = blockIdx.x * blockDim.x + threadIdx.x;
    if (i >= M_ * D_) return;
    float v = x[i] + o[i];
    x[i]  = v;
    xb[i] = f2bf(v);
}

__global__ void k_layernorm(__bf16* __restrict__ xn, const float* __restrict__ x,
                            const float* __restrict__ g, const float* __restrict__ bta) {
    __shared__ float red[256];
    int row = blockIdx.x;
    const float* xr = x + (size_t)row * D_;
    float s = 0.f;
    for (int i = threadIdx.x; i < D_; i += 256) s += xr[i];
    red[threadIdx.x] = s; __syncthreads();
    for (int st = 128; st > 0; st >>= 1) {
        if (threadIdx.x < st) red[threadIdx.x] += red[threadIdx.x + st];
        __syncthreads();
    }
    float mu = red[0] * (1.f / D_);
    __syncthreads();
    float v = 0.f;
    for (int i = threadIdx.x; i < D_; i += 256) { float d = xr[i] - mu; v += d * d; }
    red[threadIdx.x] = v; __syncthreads();
    for (int st = 128; st > 0; st >>= 1) {
        if (threadIdx.x < st) red[threadIdx.x] += red[threadIdx.x + st];
        __syncthreads();
    }
    float rstd = rsqrtf(red[0] * (1.f / D_) + 1e-5f);
    for (int i = threadIdx.x; i < D_; i += 256)
        xn[(size_t)row * D_ + i] = f2bf((xr[i] - mu) * rstd * g[i] + bta[i]);
}

__global__ void k_loss_row(float* __restrict__ nll, const float* __restrict__ logits,
                           const int* __restrict__ tgt) {
    __shared__ float red[256];
    int row = blockIdx.x;
    const float* lr = logits + (size_t)row * V_;
    float m = -1e30f;
    for (int i = threadIdx.x; i < V_; i += 256) m = fmaxf(m, lr[i]);
    red[threadIdx.x] = m; __syncthreads();
    for (int st = 128; st > 0; st >>= 1) {
        if (threadIdx.x < st) red[threadIdx.x] = fmaxf(red[threadIdx.x], red[threadIdx.x + st]);
        __syncthreads();
    }
    m = red[0]; __syncthreads();
    float sum = 0.f;
    for (int i = threadIdx.x; i < V_; i += 256) sum += __expf(lr[i] - m);
    red[threadIdx.x] = sum; __syncthreads();
    for (int st = 128; st > 0; st >>= 1) {
        if (threadIdx.x < st) red[threadIdx.x] += red[threadIdx.x + st];
        __syncthreads();
    }
    if (threadIdx.x == 0) {
        int t = tgt[row];
        bool valid = (t != -100);
        int ti = valid ? t : 0;
        float lp = lr[ti] - m - __logf(red[0]);
        nll[row] = valid ? -lp : 0.f;
    }
}

__global__ void k_loss_final(float* __restrict__ loss, const float* __restrict__ nll,
                             const int* __restrict__ tgt) {
    __shared__ float rs[256], rc[256];
    float s = 0.f, c = 0.f;
    for (int i = threadIdx.x; i < M_; i += 256) {
        s += nll[i];
        c += (tgt[i] != -100) ? 1.f : 0.f;
    }
    rs[threadIdx.x] = s; rc[threadIdx.x] = c; __syncthreads();
    for (int st = 128; st > 0; st >>= 1) {
        if (threadIdx.x < st) { rs[threadIdx.x] += rs[threadIdx.x + st]; rc[threadIdx.x] += rc[threadIdx.x + st]; }
        __syncthreads();
    }
    if (threadIdx.x == 0) *loss = rs[0] / fmaxf(rc[0], 1.f);
}

// ---------------------------------------------------------------------------
// Host orchestration
// ---------------------------------------------------------------------------
extern "C" void kernel_launch(void* const* d_in, const int* in_sizes, int n_in,
                              void* d_out, int out_size, void* d_ws, size_t ws_size,
                              hipStream_t stream) {
    const int*   ids   = (const int*)  d_in[0];
    const int*   tgt   = (const int*)  d_in[1];
    const float* tok   = (const float*)d_in[2];
    const float* pos   = (const float*)d_in[3];
    const float* ln_g  = (const float*)d_in[4];
    const float* ln_b  = (const float*)d_in[5];
    const float* headw = (const float*)d_in[6];
    const float* ipw   = (const float*)d_in[7];
    const float* cw    = (const float*)d_in[8];
    const float* cb    = (const float*)d_in[9];
    const float* xpw   = (const float*)d_in[10];
    const float* dtw   = (const float*)d_in[11];
    const float* dtbp  = (const float*)d_in[12];
    const float* Alog  = (const float*)d_in[13];
    const float* Dp    = (const float*)d_in[14];
    const float* opw   = (const float*)d_in[15];

    float* logits = (float*)d_out;                       // [M_, V_]
    float* loss   = logits + (size_t)M_ * V_;            // scalar

    // Workspace carve-up (256 B aligned)
    char* base = (char*)d_ws;
    size_t off = 0;
    auto alloc = [&](size_t bytes) -> char* {
        off = (off + 255) & ~(size_t)255;
        char* p = base + off;
        off += bytes;
        return p;
    };
    __bf16* wb    = (__bf16*)alloc((size_t)2 * E_ * D_ * 2);     // reused weight buf (max: in_proj)
    __bf16* wh    = (__bf16*)alloc((size_t)V_ * D_ * 2);         // head weight bf16
    float*  x     = (float*) alloc((size_t)M_ * D_ * 4);
    __bf16* xb    = (__bf16*)alloc((size_t)M_ * D_ * 2);
    float*  xz    = (float*) alloc((size_t)M_ * 2 * E_ * 4);
    float*  xc    = (float*) alloc((size_t)M_ * E_ * 4);
    __bf16* xcb   = (__bf16*)alloc((size_t)M_ * E_ * 2);
    float*  xdbl  = (float*) alloc((size_t)M_ * (R_ + 2 * N_) * 4);
    __bf16* dtb16 = (__bf16*)alloc((size_t)M_ * R_ * 2);
    float*  dpre  = (float*) alloc((size_t)M_ * E_ * 4);
    float*  ycore = (float*) alloc((size_t)M_ * E_ * 4);
    __bf16* yb    = (__bf16*)alloc((size_t)M_ * E_ * 2);
    float*  o     = (float*) alloc((size_t)M_ * D_ * 4);
    __bf16* xnb   = (__bf16*)alloc((size_t)M_ * D_ * 2);
    float*  nll   = (float*) alloc((size_t)M_ * 4);
    (void)ws_size; (void)n_in; (void)in_sizes; (void)out_size;

    const int TB = 256;
    auto blocks = [](size_t n, int tb) { return (unsigned)((n + tb - 1) / tb); };

    // Head weight -> bf16 (once)
    k_cvt_bf16<<<blocks((size_t)V_ * D_, TB), TB, 0, stream>>>(wh, headw, V_ * D_);

    // Embedding
    k_embed<<<blocks((size_t)M_ * D_, TB), TB, 0, stream>>>(x, xb, ids, tok, pos);

    for (int l = 0; l < L_; ++l) {
        // in_proj: xz = x @ ipw[l].T   [2048 x 4096 x 1024]
        k_cvt_bf16<<<blocks((size_t)2 * E_ * D_, TB), TB, 0, stream>>>(wb, ipw + (size_t)l * 2 * E_ * D_, 2 * E_ * D_);
        k_gemm_bf16<<<dim3((2 * E_ + BN - 1) / BN, M_ / BM), TB, 0, stream>>>(xz, xb, wb, M_, 2 * E_, D_);

        // depthwise conv + SiLU
        k_conv_silu<<<blocks((size_t)M_ * E_, TB), TB, 0, stream>>>(xc, xcb, xz, cw + (size_t)l * E_ * K_, cb + (size_t)l * E_);

        // x_proj: xdbl = xc @ xpw[l].T   [2048 x 96 x 2048]
        k_cvt_bf16<<<blocks((size_t)(R_ + 2 * N_) * E_, TB), TB, 0, stream>>>(wb, xpw + (size_t)l * (R_ + 2 * N_) * E_, (R_ + 2 * N_) * E_);
        k_gemm_bf16<<<dim3((R_ + 2 * N_ + BN - 1) / BN, M_ / BM), TB, 0, stream>>>(xdbl, xcb, wb, M_, R_ + 2 * N_, E_);

        // dt_proj: dpre = dt @ dtw[l].T   [2048 x 2048 x 64]
        k_extract_dt<<<blocks((size_t)M_ * R_, TB), TB, 0, stream>>>(dtb16, xdbl);
        k_cvt_bf16<<<blocks((size_t)E_ * R_, TB), TB, 0, stream>>>(wb, dtw + (size_t)l * E_ * R_, E_ * R_);
        k_gemm_bf16<<<dim3(E_ / BN, M_ / BM), TB, 0, stream>>>(dpre, dtb16, wb, M_, E_, R_);

        // selective scan
        k_scan<<<blocks((size_t)B_ * E_, TB), TB, 0, stream>>>(ycore, dpre, xdbl, xc,
                                                               Alog + (size_t)l * E_ * N_,
                                                               dtbp + (size_t)l * E_);
        // gate
        k_gate<<<blocks((size_t)M_ * E_, TB), TB, 0, stream>>>(yb, ycore, xc, xz, Dp + (size_t)l * E_);

        // out_proj + residual: x += y @ opw[l].T   [2048 x 1024 x 2048]
        k_cvt_bf16<<<blocks((size_t)D_ * E_, TB), TB, 0, stream>>>(wb, opw + (size_t)l * D_ * E_, D_ * E_);
        k_gemm_bf16<<<dim3(D_ / BN, M_ / BM), TB, 0, stream>>>(o, yb, wb, M_, D_, E_);
        k_residual<<<blocks((size_t)M_ * D_, TB), TB, 0, stream>>>(x, xb, o);
    }

    // Final LayerNorm -> bf16
    k_layernorm<<<M_, TB, 0, stream>>>(xnb, x, ln_g, ln_b);

    // LM head: logits = xn @ head_w.T   [2048 x 32000 x 1024]
    k_gemm_bf16<<<dim3(V_ / BN, M_ / BM), TB, 0, stream>>>(logits, xnb, wh, M_, V_, D_);

    // Cross-entropy (deterministic two-stage reduction)
    k_loss_row<<<M_, TB, 0, stream>>>(nll, logits, tgt);
    k_loss_final<<<1, TB, 0, stream>>>(loss, nll, tgt);
}
